// HeteroGNN_10720238371046
// MI455X (gfx1250) — compile-verified, matrix operand order checked by reading, not run on hardware
//
#include <hip/hip_runtime.h>

// ---------------------------------------------------------------------------
// HeteroGNN forward for MI455X (gfx1250, wave32, WMMA).
//
// Input order (setup_inputs dict insertion order, params flattened in
// insertion order):
//  0 x_item[NI*256] f32   1 x_user[NU*256] f32   2 y_emb[NI] i32
//  3 ei_iu[2*E] i32       4 ei_ui[2*E] i32       5 ei_ii[2*E] i32
//  6 emb_w[3*256]
//  7 Wr0_iu  8 br0_iu  9 Ws0_iu   10 Wr0_ui 11 br0_ui 12 Ws0_ui
// 13 Wr0_ii 14 br0_ii 15 Ws0_ii
// 16..19 bn0_item_{g,b,m,v}   20..23 bn0_user_{g,b,m,v}
// 24 Wr1_iu 25 br1_iu 26 Ws1_iu 27 Wr1_ui 28 br1_ui 29 Ws1_ui
// 30 Wr1_ii 31 br1_ii 32 Ws1_ii
// 33..36 bn1_item_{g,b,m,v}   37..40 bn1_user_{g,b,m,v}
// 41 W1[1280*128] 42 b1[128]  43..46 bnf_{g,b,m,v}[128]
// 47 W2[128*2] 48 b2[2]
// ---------------------------------------------------------------------------

typedef __attribute__((ext_vector_type(16))) __bf16 v16bf;
typedef __attribute__((ext_vector_type(8)))  __bf16 v8bf;
typedef __attribute__((ext_vector_type(8)))  float  v8f;

static constexpr int NI = 100000;
static constexpr int NU = 100000;
static constexpr int D_IN = 256;
static constexpr int H_DIM = 512;
static constexpr int N_EDGE = 800000;

// ------------------------------ small helpers ------------------------------

__global__ void k_zero_f32(float* __restrict__ p, long long n) {
  long long i = (long long)blockIdx.x * blockDim.x + threadIdx.x;
  if (i < n) p[i] = 0.0f;
}

__global__ void k_f32_to_bf16(const float* __restrict__ src,
                              __bf16* __restrict__ dst, long long n) {
  long long i = (long long)blockIdx.x * blockDim.x + threadIdx.x;
  if (i < n) dst[i] = (__bf16)src[i];
}

// xi0 = x_item + emb_w[y] * (y != 2), stored bf16
__global__ void k_embed_items(const float* __restrict__ x,
                              const int* __restrict__ y,
                              const float* __restrict__ emb,
                              __bf16* __restrict__ out) {
  long long i = (long long)blockIdx.x * blockDim.x + threadIdx.x;
  if (i >= (long long)NI * D_IN) return;
  int row = (int)(i >> 8);   // D_IN == 256
  int c   = (int)(i & 255);
  int yy  = y[row];
  float v = x[i];
  if (yy != 2) v += emb[yy * D_IN + c];
  out[i] = (__bf16)v;
}

// BN(eval) fold: scale = g*rsqrt(v+eps); shift = (bias - m)*scale + b
__global__ void k_fold_bn(const float* __restrict__ g, const float* __restrict__ b,
                          const float* __restrict__ m, const float* __restrict__ v,
                          const float* __restrict__ bias0,
                          const float* __restrict__ bias1, int useB1, int n,
                          float* __restrict__ scale, float* __restrict__ shift) {
  int i = blockIdx.x * blockDim.x + threadIdx.x;
  if (i >= n) return;
  float s = g[i] * rsqrtf(v[i] + 1e-5f);
  float bias = bias0[i] + (useB1 ? bias1[i] : 0.0f);
  scale[i] = s;
  shift[i] = (bias - m[i]) * s + b[i];
}

// ------------------------- weight fragment packing -------------------------
// Pack one K-segment of a row-major [Ksrc, N] f32 weight into the WMMA B-frag
// layout: dst[((ntile*nKt + ktOff+ktl)*32 + lane)*16 + e], where element e of
// lane covers k = ktl*32 + (e>>3)*16 + (lane>>4)*8 + (e&7), n = ntile*16+lane&15.
__global__ void k_pack_weight(const float* __restrict__ W,
                              const float* __restrict__ Wadd, int useAdd,
                              int Ksrc, int N,
                              __bf16* __restrict__ dst, int ktOff, int nKt) {
  long long tid = (long long)blockIdx.x * blockDim.x + threadIdx.x;
  long long total = (long long)Ksrc * N;
  if (tid >= total) return;
  int e    = (int)(tid & 15);
  int lane = (int)((tid >> 4) & 31);
  long long tileIdx = tid >> 9;
  int nKl   = Ksrc >> 5;
  int ktl   = (int)(tileIdx % nKl);
  int ntile = (int)(tileIdx / nKl);
  int k = (ktl << 5) + ((e >> 3) << 4) + ((lane >> 4) << 3) + (e & 7);
  int n = (ntile << 4) + (lane & 15);
  float v = W[(long long)k * N + n];
  if (useAdd) v += Wadd[(long long)k * N + n];
  dst[(((long long)ntile * nKt + (ktOff + ktl)) * 32 + lane) * 16 + e] = (__bf16)v;
}

// ----------------------------- edge scatter-add ----------------------------
// agg[dst] += src_feat[src], accumulated directly in bf16 with packed
// GLOBAL_ATOMIC_PK_ADD_BF16 (2 channels / atomic). One wave per edge:
// each lane gathers 8 bf16 via one b128 load and issues 4 packed atomics per
// 256-channel chunk. Agg buffers (bf16) are small enough to stay essentially
// resident in the 192MB global L2, where the atomics resolve.
__global__ __launch_bounds__(256)
void k_scatter_pk(const __bf16* __restrict__ src,
                  const int* __restrict__ ei, int E, int din,
                  __bf16* __restrict__ dst) {
  const int lane = threadIdx.x & 31;
  const int widx = threadIdx.x >> 5;
  const long long e = (long long)blockIdx.x * (blockDim.x >> 5) + widx;
  if (e >= E) return;
  const int s = ei[e];
  const int d = ei[E + e];
  const __bf16* sp = src + (long long)s * din;
  __bf16* dp = dst + (long long)d * din;
  if (lane == 0) __builtin_prefetch(sp, 0, 0);  // global_prefetch_b8
  const int nIter = din >> 8;  // 256 channels per wave-iteration
#pragma unroll
  for (int it = 0; it < 2; ++it) {
    if (it >= nIter) break;
    const int off = (it << 8) + (lane << 3);       // 8 bf16 per lane
    const uint4 v = *(const uint4*)(sp + off);     // global_load_b128 gather
    unsigned long long a0 = (unsigned long long)(uintptr_t)(dp + off);
    asm volatile("global_atomic_pk_add_bf16 %0, %1, off"
                 :: "v"(a0), "v"(v.x) : "memory");
    asm volatile("global_atomic_pk_add_bf16 %0, %1, off"
                 :: "v"(a0 + 4ull), "v"(v.y) : "memory");
    asm volatile("global_atomic_pk_add_bf16 %0, %1, off"
                 :: "v"(a0 + 8ull), "v"(v.z) : "memory");
    asm volatile("global_atomic_pk_add_bf16 %0, %1, off"
                 :: "v"(a0 + 12ull), "v"(v.w) : "memory");
  }
}

// --------------------------- fused WMMA GEMM -------------------------------
// C[M,N] = relu( (sum_seg A_seg @ Wseg) * scale + shift ), bf16 in / bf16 out,
// f32 accumulate via v_wmma_f32_16x16x32_bf16.
// Block: 256 thr = 8 waves; block tile 128M x 128N; wave tile 32M x 64N
// (2x4 grid of 16x16 WMMA tiles -> 8 wmma per K-step of 32).
__global__ __launch_bounds__(256)
void k_gemm_bf16_fused(const __bf16* __restrict__ A0, int K0,
                       const __bf16* __restrict__ A1, int K1,
                       const __bf16* __restrict__ A2, int K2,
                       const __bf16* __restrict__ Wp,
                       const float* __restrict__ scale,
                       const float* __restrict__ shift,
                       __bf16* __restrict__ Cout, int M, int N, int doRelu) {
  const int tid  = threadIdx.x;
  const int lane = tid & 31;
  const int wid  = tid >> 5;      // 0..7
  const int wm   = wid & 3;       // 4 waves along M
  const int wn   = wid >> 2;      // 2 waves along N
  const int rowBase = blockIdx.x * 128 + wm * 32;
  const int colBase = blockIdx.y * 128 + wn * 64;
  const int Ktot = K0 + K1 + K2;
  const int nKt  = Ktot >> 5;
  const int lm = lane & 15;       // lane % 16
  const int lh = lane >> 4;       // K-half select

  v8f acc[2][4];
#pragma unroll
  for (int mi = 0; mi < 2; ++mi)
#pragma unroll
    for (int ni = 0; ni < 4; ++ni)
#pragma unroll
      for (int r = 0; r < 8; ++r) acc[mi][ni][r] = 0.0f;

  const int ntile0 = colBase >> 4;

  for (int kt = 0; kt < nKt; ++kt) {
    const int k0 = kt << 5;
    const __bf16* Ap;
    int lk, ld;
    if (k0 < K0)            { Ap = A0; lk = k0;           ld = K0; }
    else if (k0 < K0 + K1)  { Ap = A1; lk = k0 - K0;      ld = K1; }
    else                    { Ap = A2; lk = k0 - K0 - K1; ld = K2; }

    // A fragments: row-major bf16 activations; per-lane two 16B loads.
    v16bf afr[2];
#pragma unroll
    for (int mi = 0; mi < 2; ++mi) {
      int row = rowBase + mi * 16 + lm;
      row = row < M ? row : M - 1;
      const __bf16* p = Ap + (long long)row * ld + lk + lh * 8;
      v8bf lo = *(const v8bf*)p;
      v8bf hi = *(const v8bf*)(p + 16);
#pragma unroll
      for (int i = 0; i < 8; ++i) { afr[mi][i] = lo[i]; afr[mi][i + 8] = hi[i]; }
    }

    // B fragments from pre-packed layout: per-lane 32 contiguous bytes.
    v16bf bfr[4];
#pragma unroll
    for (int ni = 0; ni < 4; ++ni) {
      const __bf16* p =
          Wp + (((long long)(ntile0 + ni) * nKt + kt) * 32 + lane) * 16;
      v8bf lo = *(const v8bf*)p;
      v8bf hi = *(const v8bf*)(p + 8);
#pragma unroll
      for (int i = 0; i < 8; ++i) { bfr[ni][i] = lo[i]; bfr[ni][i + 8] = hi[i]; }
    }

#pragma unroll
    for (int mi = 0; mi < 2; ++mi)
#pragma unroll
      for (int ni = 0; ni < 4; ++ni)
        acc[mi][ni] = __builtin_amdgcn_wmma_f32_16x16x32_bf16(
            false, afr[mi], false, bfr[ni], (short)0, acc[mi][ni], false, false);
  }

  // Epilogue: per-channel scale/shift (folded BN+bias), optional ReLU, bf16 out.
  // C-layout: VGPR r, lanes0-15 -> M=r, lanes16-31 -> M=r+8; N = lane%16.
#pragma unroll
  for (int mi = 0; mi < 2; ++mi)
#pragma unroll
    for (int ni = 0; ni < 4; ++ni) {
      int col = colBase + ni * 16 + lm;
      float sc = scale[col], sh = shift[col];
#pragma unroll
      for (int r = 0; r < 8; ++r) {
        int row = rowBase + mi * 16 + lh * 8 + r;
        if (row < M) {
          float v = acc[mi][ni][r] * sc + sh;
          if (doRelu) v = v > 0.0f ? v : 0.0f;
          Cout[(long long)row * N + col] = (__bf16)v;
        }
      }
    }
}

// ------------------------------- final head --------------------------------
// out[i,:2] = hf[i,:128] @ W2 + b2  (N=2: plain VALU dot, f32 out)
__global__ void k_final_head(const __bf16* __restrict__ hf,
                             const float* __restrict__ W2,
                             const float* __restrict__ b2,
                             float* __restrict__ out) {
  int i = blockIdx.x * blockDim.x + threadIdx.x;
  if (i >= NI) return;
  float a0 = b2[0], a1 = b2[1];
  const __bf16* h = hf + (long long)i * 128;
#pragma unroll
  for (int c8 = 0; c8 < 16; ++c8) {
    v8bf hv = *(const v8bf*)(h + c8 * 8);   // 16B activation chunks
#pragma unroll
    for (int j = 0; j < 8; ++j) {
      int c = c8 * 8 + j;
      float x = (float)hv[j];
      a0 += x * W2[c * 2 + 0];
      a1 += x * W2[c * 2 + 1];
    }
  }
  out[i * 2 + 0] = a0;
  out[i * 2 + 1] = a1;
}

// ------------------------------- host driver -------------------------------

static inline char* bump(char*& p, size_t bytes) {
  char* r = p;
  p += (bytes + 255) & ~(size_t)255;
  return r;
}

extern "C" void kernel_launch(void* const* d_in, const int* in_sizes, int n_in,
                              void* d_out, int out_size, void* d_ws, size_t ws_size,
                              hipStream_t stream) {
  (void)in_sizes; (void)n_in; (void)out_size; (void)ws_size;

  const float* x_item = (const float*)d_in[0];
  const float* x_user = (const float*)d_in[1];
  const int*   y_emb  = (const int*)d_in[2];
  const int*   ei_iu  = (const int*)d_in[3];
  const int*   ei_ui  = (const int*)d_in[4];
  const int*   ei_ii  = (const int*)d_in[5];
  const float* emb_w  = (const float*)d_in[6];
  const float* Wr0_iu = (const float*)d_in[7],  *br0_iu = (const float*)d_in[8],  *Ws0_iu = (const float*)d_in[9];
  const float* Wr0_ui = (const float*)d_in[10], *br0_ui = (const float*)d_in[11], *Ws0_ui = (const float*)d_in[12];
  const float* Wr0_ii = (const float*)d_in[13], *br0_ii = (const float*)d_in[14], *Ws0_ii = (const float*)d_in[15];
  const float* bn0i_g = (const float*)d_in[16], *bn0i_b = (const float*)d_in[17], *bn0i_m = (const float*)d_in[18], *bn0i_v = (const float*)d_in[19];
  const float* bn0u_g = (const float*)d_in[20], *bn0u_b = (const float*)d_in[21], *bn0u_m = (const float*)d_in[22], *bn0u_v = (const float*)d_in[23];
  const float* Wr1_ui = (const float*)d_in[27], *br1_ui = (const float*)d_in[28], *Ws1_ui = (const float*)d_in[29];
  const float* Wr1_ii = (const float*)d_in[30], *br1_ii = (const float*)d_in[31], *Ws1_ii = (const float*)d_in[32];
  const float* bn1i_g = (const float*)d_in[33], *bn1i_b = (const float*)d_in[34], *bn1i_m = (const float*)d_in[35], *bn1i_v = (const float*)d_in[36];
  const float* W1     = (const float*)d_in[41], *b1     = (const float*)d_in[42];
  const float* bnf_g  = (const float*)d_in[43], *bnf_b  = (const float*)d_in[44], *bnf_m  = (const float*)d_in[45], *bnf_v  = (const float*)d_in[46];
  const float* W2     = (const float*)d_in[47], *b2     = (const float*)d_in[48];
  float* out = (float*)d_out;

  // -------- workspace layout (bump allocated) --------
  char* p = (char*)d_ws;
  __bf16* xi0_bf  = (__bf16*)bump(p, (size_t)NI * D_IN * 2);
  __bf16* xu0_bf  = (__bf16*)bump(p, (size_t)NU * D_IN * 2);
  __bf16* xi1_bf  = (__bf16*)bump(p, (size_t)NI * H_DIM * 2);
  __bf16* xu1_bf  = (__bf16*)bump(p, (size_t)NU * H_DIM * 2);
  __bf16* xi2_bf  = (__bf16*)bump(p, (size_t)NI * H_DIM * 2);
  __bf16* hf_bf   = (__bf16*)bump(p, (size_t)NI * 128 * 2);
  __bf16* aggU_bf  = (__bf16*)bump(p, (size_t)NU * H_DIM * 2);  // bf16 atomic accum
  __bf16* aggI1_bf = (__bf16*)bump(p, (size_t)NI * H_DIM * 2);
  __bf16* aggI2_bf = (__bf16*)bump(p, (size_t)NI * H_DIM * 2);
  __bf16* P_u0 = (__bf16*)bump(p, (size_t)512 * 512 * 2);   // [Wr0_iu; Ws0_iu]
  __bf16* P_i0 = (__bf16*)bump(p, (size_t)768 * 512 * 2);   // [Wr0_ui; Wr0_ii; Ws0_ui+Ws0_ii]
  __bf16* P_i1 = (__bf16*)bump(p, (size_t)1536 * 512 * 2);  // [Wr1_ui; Wr1_ii; Ws1_ui+Ws1_ii]
  __bf16* P_jk = (__bf16*)bump(p, (size_t)1280 * 128 * 2);  // W1
  float* sc_u0 = (float*)bump(p, 512 * 4); float* sh_u0 = (float*)bump(p, 512 * 4);
  float* sc_i0 = (float*)bump(p, 512 * 4); float* sh_i0 = (float*)bump(p, 512 * 4);
  float* sc_i1 = (float*)bump(p, 512 * 4); float* sh_i1 = (float*)bump(p, 512 * 4);
  float* sc_jk = (float*)bump(p, 128 * 4); float* sh_jk = (float*)bump(p, 128 * 4);

  const int T = 256;
  auto blks = [](long long n, int t) { return (unsigned)((n + t - 1) / t); };

  // -------- BN/bias folding --------
  k_fold_bn<<<2, T, 0, stream>>>(bn0u_g, bn0u_b, bn0u_m, bn0u_v, br0_iu, br0_iu, 0, 512, sc_u0, sh_u0);
  k_fold_bn<<<2, T, 0, stream>>>(bn0i_g, bn0i_b, bn0i_m, bn0i_v, br0_ui, br0_ii, 1, 512, sc_i0, sh_i0);
  k_fold_bn<<<2, T, 0, stream>>>(bn1i_g, bn1i_b, bn1i_m, bn1i_v, br1_ui, br1_ii, 1, 512, sc_i1, sh_i1);
  k_fold_bn<<<1, 128, 0, stream>>>(bnf_g, bnf_b, bnf_m, bnf_v, b1, b1, 0, 128, sc_jk, sh_jk);

  // -------- weight packing into WMMA B-fragment layout --------
  k_pack_weight<<<blks((long long)256 * 512, T), T, 0, stream>>>(Wr0_iu, Wr0_iu, 0, 256, 512, P_u0, 0, 16);
  k_pack_weight<<<blks((long long)256 * 512, T), T, 0, stream>>>(Ws0_iu, Ws0_iu, 0, 256, 512, P_u0, 8, 16);
  k_pack_weight<<<blks((long long)256 * 512, T), T, 0, stream>>>(Wr0_ui, Wr0_ui, 0, 256, 512, P_i0, 0, 24);
  k_pack_weight<<<blks((long long)256 * 512, T), T, 0, stream>>>(Wr0_ii, Wr0_ii, 0, 256, 512, P_i0, 8, 24);
  k_pack_weight<<<blks((long long)256 * 512, T), T, 0, stream>>>(Ws0_ui, Ws0_ii, 1, 256, 512, P_i0, 16, 24);
  k_pack_weight<<<blks((long long)512 * 512, T), T, 0, stream>>>(Wr1_ui, Wr1_ui, 0, 512, 512, P_i1, 0, 48);
  k_pack_weight<<<blks((long long)512 * 512, T), T, 0, stream>>>(Wr1_ii, Wr1_ii, 0, 512, 512, P_i1, 16, 48);
  k_pack_weight<<<blks((long long)512 * 512, T), T, 0, stream>>>(Ws1_ui, Ws1_ii, 1, 512, 512, P_i1, 32, 48);
  k_pack_weight<<<blks((long long)1280 * 128, T), T, 0, stream>>>(W1, W1, 0, 1280, 128, P_jk, 0, 40);

  // -------- input embedding + bf16 conversion --------
  k_embed_items<<<blks((long long)NI * D_IN, T), T, 0, stream>>>(x_item, y_emb, emb_w, xi0_bf);
  k_f32_to_bf16<<<blks((long long)NU * D_IN, T), T, 0, stream>>>(x_user, xu0_bf, (long long)NU * D_IN);

  // ================= layer 0 =================
  // zero bf16 agg regions (D_IN columns used): write 0x0000_0000 dwords.
  k_zero_f32<<<blks((long long)NU * D_IN / 2, T), T, 0, stream>>>((float*)aggU_bf,  (long long)NU * D_IN / 2);
  k_zero_f32<<<blks((long long)NI * D_IN / 2, T), T, 0, stream>>>((float*)aggI1_bf, (long long)NI * D_IN / 2);
  k_zero_f32<<<blks((long long)NI * D_IN / 2, T), T, 0, stream>>>((float*)aggI2_bf, (long long)NI * D_IN / 2);
  k_scatter_pk<<<(N_EDGE + 7) / 8, T, 0, stream>>>(xi0_bf, ei_iu, N_EDGE, D_IN, aggU_bf);
  k_scatter_pk<<<(N_EDGE + 7) / 8, T, 0, stream>>>(xu0_bf, ei_ui, N_EDGE, D_IN, aggI1_bf);
  k_scatter_pk<<<(N_EDGE + 7) / 8, T, 0, stream>>>(xi0_bf, ei_ii, N_EDGE, D_IN, aggI2_bf);

  dim3 gU((NU + 127) / 128, 512 / 128);
  dim3 gI((NI + 127) / 128, 512 / 128);
  // xu1 = relu(bn([aggU | xu0] @ [Wr0_iu; Ws0_iu] + br0_iu))
  k_gemm_bf16_fused<<<gU, T, 0, stream>>>(aggU_bf, 256, xu0_bf, 256, xu0_bf, 0,
                                          P_u0, sc_u0, sh_u0, xu1_bf, NU, 512, 1);
  // xi1 = relu(bn([aggI1 | aggI2 | xi0] @ [Wr0_ui; Wr0_ii; Ws0_ui+Ws0_ii] + br))
  k_gemm_bf16_fused<<<gI, T, 0, stream>>>(aggI1_bf, 256, aggI2_bf, 256, xi0_bf, 256,
                                          P_i0, sc_i0, sh_i0, xi1_bf, NI, 512, 1);

  // ================= layer 1 (item branch only; xu2 is dead) =================
  k_zero_f32<<<blks((long long)NI * H_DIM / 2, T), T, 0, stream>>>((float*)aggI1_bf, (long long)NI * H_DIM / 2);
  k_zero_f32<<<blks((long long)NI * H_DIM / 2, T), T, 0, stream>>>((float*)aggI2_bf, (long long)NI * H_DIM / 2);
  k_scatter_pk<<<(N_EDGE + 7) / 8, T, 0, stream>>>(xu1_bf, ei_ui, N_EDGE, H_DIM, aggI1_bf);
  k_scatter_pk<<<(N_EDGE + 7) / 8, T, 0, stream>>>(xi1_bf, ei_ii, N_EDGE, H_DIM, aggI2_bf);

  k_gemm_bf16_fused<<<gI, T, 0, stream>>>(aggI1_bf, 512, aggI2_bf, 512, xi1_bf, 512,
                                          P_i1, sc_i1, sh_i1, xi2_bf, NI, 512, 1);

  // ================= JK-concat GEMM + BN + ReLU =================
  dim3 gJ((NI + 127) / 128, 1);
  k_gemm_bf16_fused<<<gJ, T, 0, stream>>>(xi0_bf, 256, xi1_bf, 512, xi2_bf, 512,
                                          P_jk, sc_jk, sh_jk, hf_bf, NI, 128, 1);

  // ================= final 128 -> 2 head =================
  k_final_head<<<blks(NI, T), T, 0, stream>>>(hf_bf, W2, b2, out);
}
